// MaxPool_730144440853
// MI455X (gfx1250) — matrix operand order repaired and checked by reading, not compiled, not run on hardware
//
#include <hip/hip_runtime.h>
#include <hip/hip_bf16.h>

// ---------------------------------------------------------------------------
// Problem: B=8, C=256, Z=3, N=8192
//   d = einsum("bczn,dc->bdzn", x, W); scores = sum_z x*d; idx = argmax_n;
//   out[b,c,:] = x[b,c,:,idx[b,c]]
// Strategy: bf16 WMMA (f32 accum) fused GEMM + diag-score + argmax, 64-bit
// atomicMax merge of (sortable_score, ~n), fp32 gather at the end.
// ---------------------------------------------------------------------------

typedef __attribute__((ext_vector_type(16))) __bf16        v16bf;
typedef __attribute__((ext_vector_type(8)))  float         v8f;
typedef __attribute__((ext_vector_type(8)))  unsigned int  v8u;

#define BATCH 8
#define CH    256
#define NPT   8192
#define NT    64          // n columns per workgroup tile
#define LDC   132         // dwords per LDS column (128 used + pad, 16B aligned)

__device__ __forceinline__ unsigned f2bf_rne(float f) {
    unsigned u = __float_as_uint(f);
    unsigned r = u + 0x7FFFu + ((u >> 16) & 1u);
    return r >> 16;
}

// ---------------- prep: W fp32 -> bf16, init best[] = 0 ---------------------
__global__ void prep_kernel(const float* __restrict__ W,
                            unsigned short* __restrict__ Wbf,
                            unsigned long long* __restrict__ best) {
    int i = blockIdx.x * 256 + threadIdx.x;          // 0 .. 65535
    if (i < BATCH * CH) best[i] = 0ULL;
    if (i < CH * CH)    Wbf[i] = (unsigned short)f2bf_rne(W[i]);
}

// ---------------- main fused kernel ----------------------------------------
__global__ __launch_bounds__(256, 1)
void pool_main_kernel(const float* __restrict__ x,
                      const unsigned short* __restrict__ Wbf,
                      unsigned long long* __restrict__ best) {
    // LDS: X tiles for all 3 z, bf16 pairs packed per dword, column-major:
    // ldsX[z][n_local][c/2], dword holds c even in [15:0], c odd in [31:16].
    __shared__ __align__(16) unsigned int ldsX[3][NT][LDC];   // ~99 KB

    const int b    = blockIdx.x >> 7;        // 8 batches
    const int tile = blockIdx.x & 127;       // 128 tiles of NT=64 columns
    const int n0   = tile * NT;
    const int t    = threadIdx.x;

    // ---- stage: global fp32 -> LDS bf16 packed (coalesced n, transposed) --
    {
        const int nl  = t & 63;              // local column
        const int cp0 = t >> 6;              // 0..3 starting c-pair
        for (int z = 0; z < 3; ++z) {
            const float* xz = x + ((size_t)(b * CH) * 3 + z) * NPT + n0 + nl;
            for (int cp = cp0; cp < CH / 2; cp += 4) {
                const size_t coff = (size_t)(2 * cp) * 3 * NPT;
                float f0 = xz[coff];
                float f1 = xz[coff + (size_t)3 * NPT];
                ldsX[z][nl][cp] = f2bf_rne(f0) | (f2bf_rne(f1) << 16);
            }
        }
    }
    __syncthreads();

    const int w      = t >> 5;               // wave id 0..7
    const int l      = t & 31;               // lane
    const int colsel = l & 15;               // N / M lane select
    const int hi     = l >> 4;               // half select

    for (int rb = 0; rb < 2; ++rb) {
        const int cb = rb * 128 + w * 16;    // this wave's 16 output rows

        // ---- preload A (W rows cb..cb+15, all K=256) into registers -------
        // 16-bit A layout: lane m = cb+colsel; lanes>=16 take K chunks +8/+24.
        v16bf Areg[8];
        {
            const unsigned short* wr = Wbf + (size_t)(cb + colsel) * CH;
            #pragma unroll
            for (int k = 0; k < 8; ++k) {
                const int k0 = 32 * k + hi * 8;
                uint4 cl = *(const uint4*)(wr + k0);        // K = k0 .. k0+7
                uint4 ch = *(const uint4*)(wr + k0 + 16);   // K = k0+16 .. +23
                v8u au = {cl.x, cl.y, cl.z, cl.w, ch.x, ch.y, ch.z, ch.w};
                Areg[k] = __builtin_bit_cast(v16bf, au);
            }
        }

        float    bestS[8];
        unsigned bestN[8];
        #pragma unroll
        for (int g = 0; g < 8; ++g) { bestS[g] = -3.4e38f; bestN[g] = 0u; }

        for (int ng = 0; ng < NT / 16; ++ng) {
            const int coll = ng * 16 + colsel;      // local column this lane owns
            float sc[8] = {0.f, 0.f, 0.f, 0.f, 0.f, 0.f, 0.f, 0.f};

            for (int z = 0; z < 3; ++z) {
                v8f d = {};
                const unsigned int* colp = &ldsX[z][coll][hi * 4];
                #pragma unroll
                for (int k = 0; k < 8; ++k) {
                    const uint4* p = (const uint4*)(colp + 16 * k);
                    uint4 b0 = p[0], b1 = p[2];
                    v8u bu = {b0.x, b0.y, b0.z, b0.w, b1.x, b1.y, b1.z, b1.w};
                    v16bf Bv = __builtin_bit_cast(v16bf, bu);
                    d = __builtin_amdgcn_wmma_f32_16x16x32_bf16(
                            false, Areg[k], false, Bv, (short)0, d,
                            false, false);
                }
                // score += x[c,n] * d[c,n]; D layout: VGPR g -> M = g + 8*hi
                #pragma unroll
                for (int g = 0; g < 8; ++g) {
                    const int c  = cb + g + 8 * hi;
                    unsigned pv  = ldsX[z][coll][c >> 1];
                    unsigned xb  = (g & 1) ? (pv >> 16) : (pv & 0xFFFFu);
                    float    xf  = __uint_as_float(xb << 16);
                    sc[g] += xf * d[g];
                }
            }
            const unsigned ngl = (unsigned)(n0 + coll);
            #pragma unroll
            for (int g = 0; g < 8; ++g)
                if (sc[g] > bestS[g]) { bestS[g] = sc[g]; bestN[g] = ngl; }
        }

        // ---- reduce over 16 lanes sharing each c, then publish ------------
        #pragma unroll
        for (int g = 0; g < 8; ++g) {
            float    s = bestS[g];
            unsigned n = bestN[g];
            #pragma unroll
            for (int off = 1; off < 16; off <<= 1) {
                float    s2 = __shfl_xor(s, off, 32);
                unsigned n2 = (unsigned)__shfl_xor((int)n, off, 32);
                if (s2 > s || (s2 == s && n2 < n)) { s = s2; n = n2; }
            }
            if ((l & 15) == 0) {
                unsigned su = __float_as_uint(s);
                su = (su & 0x80000000u) ? ~su : (su | 0x80000000u);
                // ~n so ties (equal score) prefer the LOWEST n under max()
                unsigned long long packed =
                    ((unsigned long long)su << 32) | (unsigned long long)(~n);
                const int c = cb + g + 8 * hi;
                atomicMax(&best[b * CH + c], packed);
            }
        }
    }
}

// ---------------- finalize: unpack n, gather fp32 3-vector ------------------
__global__ void finalize_kernel(const float* __restrict__ x,
                                const unsigned long long* __restrict__ best,
                                float* __restrict__ out) {
    int i = blockIdx.x * 256 + threadIdx.x;          // (b*CH + c)
    if (i >= BATCH * CH) return;
    unsigned n = ~(unsigned)(best[i] & 0xFFFFFFFFull);
    n &= (NPT - 1);
    const float* xp = x + (size_t)i * 3 * NPT;
    #pragma unroll
    for (int z = 0; z < 3; ++z)
        out[i * 3 + z] = xp[(size_t)z * NPT + n];
}

// ---------------------------------------------------------------------------
extern "C" void kernel_launch(void* const* d_in, const int* in_sizes, int n_in,
                              void* d_out, int out_size, void* d_ws, size_t ws_size,
                              hipStream_t stream) {
    (void)in_sizes; (void)n_in; (void)out_size; (void)ws_size;
    const float* x = (const float*)d_in[0];   // [8,256,3,8192] fp32
    const float* W = (const float*)d_in[1];   // [256,256] fp32
    float* out = (float*)d_out;               // [8,256,3] fp32

    unsigned long long* best = (unsigned long long*)d_ws;            // 16 KB
    unsigned short*     Wbf  = (unsigned short*)((char*)d_ws + 16384); // 128 KB

    prep_kernel<<<256, 256, 0, stream>>>(W, Wbf, best);
    pool_main_kernel<<<BATCH * (NPT / NT), 256, 0, stream>>>(x, Wbf, best);
    finalize_kernel<<<(BATCH * CH + 255) / 256, 256, 0, stream>>>(x, best, out);
}
